// SHINE_4758823764218
// MI455X (gfx1250) — matrix operand order repaired and checked by reading, compiled, not run on hardware
//
#include <hip/hip_runtime.h>
#include <hip/hip_bf16.h>
#include <math.h>

// ---------------------------------------------------------------------------
// MI455X (gfx1250) implementation.
// Memory-bound pipeline (~60 GFLOP vs <1GB HBM traffic @ 23.3 TB/s).
// - All GEMMs: LDS-tiled bf16 WMMA (v_wmma_f32_16x16x32_bf16), fp32 accum.
// - 6144x6144 cosine adjacency kept in bf16 end-to-end (halves its traffic:
//   ~755MB -> ~378MB), consumed by the two final GCN layers via
//   global_load_async_to_lds_b128 DMA staging (no VGPR round trip / convert).
// ---------------------------------------------------------------------------

typedef __attribute__((ext_vector_type(4)))  float   v4f;
typedef __attribute__((ext_vector_type(8)))  float   v8f;
typedef __attribute__((ext_vector_type(4)))  __bf16  v4bf;
typedef __attribute__((ext_vector_type(8)))  __bf16  v8bf;
typedef __attribute__((ext_vector_type(16))) __bf16  v16bf;

#define BM 128
#define BN 128
#define BK 32
#define LDSS 40   // BK + 8 pad; 80 bytes/row keeps 16B alignment for b128 loads

#define GEMM_NN       0   // A fp32 [M,K], B fp32 [K,N]
#define GEMM_NT       1   // A fp32 [M,K], B fp32 [N,K]   (C = A @ B^T)
#define GEMM_NN_ABF16 2   // A bf16 [M,K] (M%128==0, K%32==0), B fp32 [K,N]

__device__ __forceinline__ __bf16 f2bf(float f) {
    union { float f; unsigned u; } a; a.f = f;
    unsigned r = a.u + 0x7FFFu + ((a.u >> 16) & 1u);   // round-to-nearest-even
    unsigned short h = (unsigned short)(r >> 16);
    __bf16 b;
    __builtin_memcpy(&b, &h, 2);
    return b;
}
__device__ __forceinline__ float bf2f(__bf16 b) {
    unsigned short h;
    __builtin_memcpy(&h, &b, 2);
    union { unsigned u; float f; } a; a.u = ((unsigned)h) << 16;
    return a.f;
}

// C[M,N] = op(A[M,K] @ B) (+bias) (relu). outbf: store C as bf16.
template<int MODE>
__global__ __launch_bounds__(256)
void gemm_bf16_kernel(const void* __restrict__ Av, const float* __restrict__ B,
                      const float* __restrict__ bias, void* __restrict__ Cv,
                      int M, int N, int K, int lda, int ldb, int ldc,
                      int relu, int outbf)
{
    __shared__ __align__(16) __bf16 As[BM * LDSS];
    __shared__ __align__(16) __bf16 Bs[BN * LDSS];

    const int tid  = threadIdx.x;
    const int lane = tid & 31;
    const int wave = tid >> 5;
    const int wm   = wave & 3;    // wave M offset = wm*32
    const int wn   = wave >> 2;   // wave N offset = wn*64
    const int m0   = blockIdx.y * BM;
    const int n0   = blockIdx.x * BN;

    v8f acc[2][4];
#pragma unroll
    for (int mi = 0; mi < 2; ++mi)
#pragma unroll
        for (int ni = 0; ni < 4; ++ni) {
            v8f z = {0.f,0.f,0.f,0.f,0.f,0.f,0.f,0.f};
            acc[mi][ni] = z;
        }

    // lane -> fragment addressing (ISA 16-bit A 16x32 layout; B per sparse doc)
    const int arow = wm * 32 + (lane & 15);
    const int akb  = ((lane >> 4) & 1) * 8;    // A K-base: 0 or 8
    const int bcol = wn * 64 + (lane & 15);
    const int bkb  = ((lane >> 4) & 1) * 8;    // B K-base: 0 or 8

    for (int k0 = 0; k0 < K; k0 += BK) {
        __syncthreads();

        // ---- stage A tile [BM][BK] bf16 into LDS ----
        if (MODE == GEMM_NN_ABF16) {
            // A already bf16; full tile in range (M%128==0, K%32==0):
            // DMA 16B chunks straight into LDS (ASYNCcnt), no VGPR round trip.
            const __bf16* Abf = (const __bf16*)Av;
#pragma unroll
            for (int c = tid; c < (BM * BK / 8); c += 256) {   // 512 x b128
                int i = c >> 2;                  // tile row
                int j = (c & 3) << 3;            // element offset (8 bf16 = 16B)
                const __bf16* gp = Abf + (size_t)(m0 + i) * lda + (k0 + j);
                unsigned ldsa = (unsigned)(size_t)&As[i * LDSS + j];
                asm volatile("global_load_async_to_lds_b128 %0, %1, off"
                             :: "v"(ldsa), "v"(gp) : "memory");
            }
        } else {
            const float* A = (const float*)Av;
            for (int s = tid; s < (BM * BK / 4); s += 256) {
                int i = s >> 3;            // tile row
                int j = (s & 7) << 2;      // tile col (mult of 4)
                int gr = m0 + i, gc = k0 + j;
                float x0 = 0.f, x1 = 0.f, x2 = 0.f, x3 = 0.f;
                if (gr < M) {
                    const float* p = A + (long)gr * lda + gc;
                    if (gc + 3 < K) { v4f v = *(const v4f*)p; x0 = v[0]; x1 = v[1]; x2 = v[2]; x3 = v[3]; }
                    else {
                        if (gc + 0 < K) x0 = p[0];
                        if (gc + 1 < K) x1 = p[1];
                        if (gc + 2 < K) x2 = p[2];
                        if (gc + 3 < K) x3 = p[3];
                    }
                }
                v4bf w; w[0] = f2bf(x0); w[1] = f2bf(x1); w[2] = f2bf(x2); w[3] = f2bf(x3);
                *(v4bf*)&As[i * LDSS + j] = w;
            }
        }

        // ---- stage B tile as [N][K] bf16 (transposed layout in LDS) ----
        if (MODE == GEMM_NT) {
            // B is [N][K] row-major -> direct copy
            for (int s = tid; s < (BN * BK / 4); s += 256) {
                int i = s >> 3;            // n index
                int j = (s & 7) << 2;      // k index
                int gn = n0 + i, gk = k0 + j;
                float x0 = 0.f, x1 = 0.f, x2 = 0.f, x3 = 0.f;
                if (gn < N) {
                    const float* p = B + (long)gn * ldb + gk;
                    if (gk + 3 < K) { v4f v = *(const v4f*)p; x0 = v[0]; x1 = v[1]; x2 = v[2]; x3 = v[3]; }
                    else {
                        if (gk + 0 < K) x0 = p[0];
                        if (gk + 1 < K) x1 = p[1];
                        if (gk + 2 < K) x2 = p[2];
                        if (gk + 3 < K) x3 = p[3];
                    }
                }
                v4bf w; w[0] = f2bf(x0); w[1] = f2bf(x1); w[2] = f2bf(x2); w[3] = f2bf(x3);
                *(v4bf*)&Bs[i * LDSS + j] = w;
            }
        } else {
            // B is [K][N] row-major -> transpose while storing
            for (int s = tid; s < (BK * BN / 4); s += 256) {
                int j = s >> 5;            // k index 0..31
                int i = (s & 31) << 2;     // n index (mult of 4)
                int gk = k0 + j, gn = n0 + i;
                float x[4] = {0.f, 0.f, 0.f, 0.f};
                if (gk < K) {
                    const float* p = B + (long)gk * ldb + gn;
                    if (gn + 3 < N) { v4f v = *(const v4f*)p; x[0] = v[0]; x[1] = v[1]; x[2] = v[2]; x[3] = v[3]; }
                    else {
#pragma unroll
                        for (int q = 0; q < 4; ++q) if (gn + q < N) x[q] = p[q];
                    }
                }
#pragma unroll
                for (int q = 0; q < 4; ++q) Bs[(i + q) * LDSS + j] = f2bf(x[q]);
            }
        }

        if (MODE == GEMM_NN_ABF16)
            asm volatile("s_wait_asynccnt 0" ::: "memory");
        __syncthreads();

        // ---- build fragments (2 x ds_load_b128 each) and issue 8 WMMAs ----
        v16bf afrag[2], bfrag[4];
#pragma unroll
        for (int mi = 0; mi < 2; ++mi) {
            const __bf16* pa = &As[(arow + mi * 16) * LDSS + akb];
            v8bf lo = *(const v8bf*)pa;
            v8bf hi = *(const v8bf*)(pa + 16);
            afrag[mi] = __builtin_shufflevector(lo, hi, 0,1,2,3,4,5,6,7,8,9,10,11,12,13,14,15);
        }
#pragma unroll
        for (int ni = 0; ni < 4; ++ni) {
            const __bf16* pb = &Bs[(bcol + ni * 16) * LDSS + bkb];
            v8bf lo = *(const v8bf*)pb;
            v8bf hi = *(const v8bf*)(pb + 16);
            bfrag[ni] = __builtin_shufflevector(lo, hi, 0,1,2,3,4,5,6,7,8,9,10,11,12,13,14,15);
        }
#pragma unroll
        for (int mi = 0; mi < 2; ++mi)
#pragma unroll
            for (int ni = 0; ni < 4; ++ni)
                acc[mi][ni] = __builtin_amdgcn_wmma_f32_16x16x32_bf16(
                    false, afrag[mi], false, bfrag[ni],
                    (short)0, acc[mi][ni], false, false);
    }

    // ---- epilogue: bias + relu + bounds-checked store (fp32 or bf16) ----
    float*  Cf = (float*)Cv;
    __bf16* Cb = (__bf16*)Cv;
    const int chalf = ((lane >> 4) & 1) * 8;  // C row half (ISA C/D layout)
    const int ccol  = lane & 15;
#pragma unroll
    for (int mi = 0; mi < 2; ++mi)
#pragma unroll
        for (int ni = 0; ni < 4; ++ni) {
            int colg = n0 + wn * 64 + ni * 16 + ccol;
            float bv = (bias && colg < N) ? bias[colg] : 0.f;
#pragma unroll
            for (int r = 0; r < 8; ++r) {
                int rowg = m0 + wm * 32 + mi * 16 + chalf + r;
                if (rowg < M && colg < N) {
                    float v = acc[mi][ni][r] + bv;
                    if (relu) v = fmaxf(v, 0.f);
                    long idx = (long)rowg * ldc + colg;
                    if (outbf) Cb[idx] = f2bf(v);
                    else       Cf[idx] = v;
                }
            }
        }
}

// In-place L2 normalize each 100-wide segment of Doc[rows][300] (one wave/segment)
__global__ __launch_bounds__(128)
void rownorm_kernel(float* __restrict__ doc, int rows)
{
    int wid  = (blockIdx.x * 128 + threadIdx.x) >> 5;
    int lane = threadIdx.x & 31;
    if (wid >= rows * 3) return;
    int row = wid / 3, seg = wid % 3;
    float* p = doc + (long)row * 300 + seg * 100;
    float s = 0.f;
    for (int i = lane; i < 100; i += 32) { float v = p[i]; s += v * v; }
#pragma unroll
    for (int off = 16; off; off >>= 1) s += __shfl_xor(s, off, 32);
    float inv = 1.f / (sqrtf(s) + 1e-9f);
    for (int i = lane; i < 100; i += 32) p[i] *= inv;
}

// In-place on bf16: A[i,:] = mask(cos)[i,:] / (rowsum + 1e-9), mask = cos*(cos>0.05)
__global__ __launch_bounds__(256)
void cosnorm_bf16_kernel(__bf16* __restrict__ cosm, int n)
{
    __shared__ float rowbuf[6144];
    __shared__ float red[256];
    int row = blockIdx.x;
    __bf16* prow = cosm + (long)row * n;
    float s = 0.f;
    for (int j = threadIdx.x; j < n; j += 256) {
        float v = bf2f(prow[j]);
        v = (v > 0.05f) ? v : 0.f;
        rowbuf[j] = v;
        s += v;
    }
    red[threadIdx.x] = s;
    __syncthreads();
    for (int off = 128; off; off >>= 1) {
        if (threadIdx.x < off) red[threadIdx.x] += red[threadIdx.x + off];
        __syncthreads();
    }
    float inv = 1.f / (red[0] + 1e-9f);
    for (int j = threadIdx.x; j < n; j += 256) prow[j] = f2bf(rowbuf[j] * inv);
}

static inline void gemm(hipStream_t st, const void* A, const float* B,
                        const float* bias, void* C,
                        int M, int N, int K, int lda, int ldb, int ldc,
                        bool relu, int mode, bool outbf = false)
{
    dim3 grid((N + BN - 1) / BN, (M + BM - 1) / BM);
    dim3 blk(256);
    if (mode == GEMM_NT)
        gemm_bf16_kernel<GEMM_NT><<<grid, blk, 0, st>>>(A, B, bias, C, M, N, K, lda, ldb, ldc, (int)relu, (int)outbf);
    else if (mode == GEMM_NN_ABF16)
        gemm_bf16_kernel<GEMM_NN_ABF16><<<grid, blk, 0, st>>>(A, B, bias, C, M, N, K, lda, ldb, ldc, (int)relu, (int)outbf);
    else
        gemm_bf16_kernel<GEMM_NN><<<grid, blk, 0, st>>>(A, B, bias, C, M, N, K, lda, ldb, ldc, (int)relu, (int)outbf);
}

extern "C" void kernel_launch(void* const* d_in, const int* in_sizes, int n_in,
                              void* d_out, int out_size, void* d_ws, size_t ws_size,
                              hipStream_t stream)
{
    const float* adj11 = (const float*)d_in[0];
    const float* adj22 = (const float*)d_in[1];
    const float* adj33 = (const float*)d_in[2];
    const float* adj01 = (const float*)d_in[3];
    const float* adj02 = (const float*)d_in[4];
    const float* adj03 = (const float*)d_in[5];
    const float* feat1 = (const float*)d_in[6];
    const float* feat2 = (const float*)d_in[7];
    const float* feat3 = (const float*)d_in[8];
    const float* W1a = (const float*)d_in[9];   const float* b1a = (const float*)d_in[10];
    const float* W1b = (const float*)d_in[11];  const float* b1b = (const float*)d_in[12];
    const float* W2a = (const float*)d_in[13];  const float* b2a = (const float*)d_in[14];
    const float* W2b = (const float*)d_in[15];  const float* b2b = (const float*)d_in[16];
    const float* W3a = (const float*)d_in[17];  const float* b3a = (const float*)d_in[18];
    const float* W3b = (const float*)d_in[19];  const float* b3b = (const float*)d_in[20];
    const float* Wr  = (const float*)d_in[21];  const float* br  = (const float*)d_in[22];
    const float* Wf  = (const float*)d_in[23];  const float* bf  = (const float*)d_in[24];
    const float* Wf2 = (const float*)d_in[25];  const float* bf2 = (const float*)d_in[26];
    const float* Wfc = (const float*)d_in[27];  const float* bfc = (const float*)d_in[28];
    float* out = (float*)d_out;

    const int n0 = 6144, n1 = 6000, n2 = 3000, n3 = 64;
    const int d1 = 300, d2 = 200, d3 = 64, h = 100, dlin = 200, C = 20;

    // workspace layout; COS region is bf16 but keeps its fp32-sized slot
    float*  ws  = (float*)d_ws;
    __bf16* COS = (__bf16*)ws;                           // n0*n0 bf16 (<= slot)
    float*  DOC = ws  + (size_t)n0 * n0;                 // n0*300
    float*  LIN = DOC + (size_t)n0 * 300;                // n0*200
    float*  H1  = LIN + (size_t)n0 * 200;                // n1*h
    float*  H2  = H1  + (size_t)n1 * h;                  // n2*h
    float*  H3  = H2  + (size_t)n2 * h;                  // n3*h
    float*  TA  = H3  + (size_t)n3 * h;                  // n1*h scratch
    float*  TB  = TA  + (size_t)n1 * h;                  // n1*h scratch
    float*  SF  = TB  + (size_t)n1 * h;                  // n0*h
    float*  F1  = SF  + (size_t)n0 * h;                  // n0*h

    // ---- Tower 1 (first layer identity: no adjacency) ----
    gemm(stream, feat1, W1a, b1a, TA, n1, h, d1, d1, h, h, true,  GEMM_NN);
    gemm(stream, TA,    W1b, 0,   TB, n1, h, h,  h,  h, h, false, GEMM_NN);
    gemm(stream, adj11, TB,  b1b, H1, n1, h, n1, n1, h, h, true,  GEMM_NN);

    // ---- Tower 2 ----
    gemm(stream, feat2, W2a, 0,   TA, n2, h, d2, d2, h, h, false, GEMM_NN);
    gemm(stream, adj22, TA,  b2a, TB, n2, h, n2, n2, h, h, true,  GEMM_NN);
    gemm(stream, TB,    W2b, 0,   TA, n2, h, h,  h,  h, h, false, GEMM_NN);
    gemm(stream, adj22, TA,  b2b, H2, n2, h, n2, n2, h, h, true,  GEMM_NN);

    // ---- Tower 3 ----
    gemm(stream, feat3, W3a, 0,   TA, n3, h, d3, d3, h, h, false, GEMM_NN);
    gemm(stream, adj33, TA,  b3a, TB, n3, h, n3, n3, h, h, true,  GEMM_NN);
    gemm(stream, TB,    W3b, 0,   TA, n3, h, h,  h,  h, h, false, GEMM_NN);
    gemm(stream, adj33, TA,  b3b, H3, n3, h, n3, n3, h, h, true,  GEMM_NN);

    // ---- Aggregate: write pooled blocks straight into concatenated Doc ----
    gemm(stream, adj01, H1, 0, DOC + 0,   n0, h, n1, n1, h, 300, false, GEMM_NN);
    gemm(stream, adj02, H2, 0, DOC + 100, n0, h, n2, n2, h, 300, false, GEMM_NN);
    gemm(stream, adj03, H3, 0, DOC + 200, n0, h, n3, n3, h, 300, false, GEMM_NN);
    {
        int waves = n0 * 3;                 // one wave per 100-wide segment
        int blocks = (waves + 3) / 4;       // 4 waves (128 threads) per block
        rownorm_kernel<<<blocks, 128, 0, stream>>>(DOC, n0);
    }

    // ---- refined linear ----
    gemm(stream, DOC, Wr, br, LIN, n0, dlin, 3 * h, 3 * h, dlin, dlin, false, GEMM_NN);

    // ---- cosine adjacency: Doc @ Doc^T stored as bf16, in-place mask+norm ----
    gemm(stream, DOC, DOC, 0, COS, n0, n0, 3 * h, 3 * h, 3 * h, n0, false, GEMM_NT, /*outbf=*/true);
    cosnorm_bf16_kernel<<<n0, 256, 0, stream>>>(COS, n0);

    // ---- final two GCN layers (bf16 adjacency via async-LDS DMA) + classifier ----
    gemm(stream, LIN, Wf,  0,   SF, n0, h, dlin, dlin, h, h, false, GEMM_NN);
    gemm(stream, COS, SF,  bf,  F1, n0, h, n0,   n0,   h, h, true,  GEMM_NN_ABF16);
    gemm(stream, F1,  Wf2, 0,   SF, n0, h, h,    h,    h, h, false, GEMM_NN);
    gemm(stream, COS, SF,  bf2, F1, n0, h, n0,   n0,   h, h, true,  GEMM_NN_ABF16);
    gemm(stream, F1,  Wfc, bfc, out, n0, C, h,   h,    C, C, false, GEMM_NN);
}
